// ColumnTransformerV2_72387378807389
// MI455X (gfx1250) — compile-verified, matrix operand order checked by reading, not run on hardware
//
#include <hip/hip_runtime.h>
#include <hip/hip_bf16.h>

// ---------------------------------------------------------------------------
// Model constants (mirror reference)
// ---------------------------------------------------------------------------
constexpr int VV   = 32000;
constexpr int DD   = 1024;
constexpr int DFF  = 4096;
constexpr int LT   = 2;
constexpr int HT   = 16;
constexpr int CCOL = 4;
constexpr int DC   = 256;
constexpr int DFFC = 1024;
constexpr int LC   = 4;
constexpr int HC   = 4;
constexpr int BB   = 2;
constexpr int TT   = 1024;
constexpr int HD   = 64;

// ---------------------------------------------------------------------------
// bf16 helpers / WMMA vector types (CDNA5 wave32 WMMA)
// ---------------------------------------------------------------------------
typedef __attribute__((ext_vector_type(16))) __bf16 v16bf;
typedef __attribute__((ext_vector_type(8)))  __bf16 v8bf;
typedef __attribute__((ext_vector_type(8)))  float  v8f;

// native f32->bf16 conversion (hardware cvt path on gfx1250)
static __device__ __forceinline__ void stage8(float4 f0, float4 f1, __bf16* dst) {
  v8f fv = {f0.x, f0.y, f0.z, f0.w, f1.x, f1.y, f1.z, f1.w};
  v8bf bv = __builtin_convertvector(fv, v8bf);
  *(v8bf*)dst = bv;
}

// epilogue flags
#define EF_BIAS 1
#define EF_RES  2
#define EF_GELU 4
#define EF_MASK 8
#define EF_NT   16

struct GemmP {
  const float* A;
  const float* Bm;
  float*       C;
  const float* bias;
  const float* res;
  int M, N, K;
  int lda, ldb, ldc;
  long long sAb, sAh, sBb, sBh, sCb, sCh;  // outer(b) / inner(h) batch strides
  long long biasSb;
  int NH;        // inner batch count (gridDim.z = NB*NH)
  int nT;        // 1 => B stored [N,K] (NT gemm), 0 => [K,N]
  float alpha;
  int flags;
};

// ---------------------------------------------------------------------------
// Generic batched GEMM:  C = epilogue(alpha * A(f32->bf16) x B(f32->bf16))
// 128x64 C-tile / 256-thread block (8 wave32s).  Each wave owns a 32x32 slab
// -> 2 A-frags x 2 B-frags -> 4x v_wmma_f32_16x16x32_bf16 per 32-deep K step.
// REQUIRES: M % 128 == 0, N % 64 == 0, K % 32 == 0 (true for every GEMM in
// this model) -- no bounds checks, so staging is straight-line b128 traffic.
// ---------------------------------------------------------------------------
#define TM 128
#define TN 64
#define LDSS 48   // 32 K-elems + 16 pad; row pitch 96B keeps b128 alignment

__global__ __launch_bounds__(256) void gemm_kernel(GemmP p) {
  __shared__ __align__(16) __bf16 As[TM * LDSS];   // 12 KB
  __shared__ __align__(16) __bf16 Bs[TN * LDSS];   //  6 KB, stored [n][k]

  const int z  = blockIdx.z;
  const int zb = z / p.NH;
  const int zh = z % p.NH;
  const float* A  = p.A  + zb * p.sAb + zh * p.sAh;
  const float* Bm = p.Bm + zb * p.sBb + zh * p.sBh;
  float*       C  = p.C  + zb * p.sCb + zh * p.sCh;
  const float* res  = p.res  ? p.res  + zb * p.sCb + zh * p.sCh : nullptr;
  const float* bias = p.bias ? p.bias + zb * p.biasSb            : nullptr;

  const int mtile = blockIdx.y * TM;
  const int ntile = blockIdx.x * TN;

  const int tid  = threadIdx.x;
  const int lane = tid & 31;
  const int wave = tid >> 5;
  const int wm   = wave >> 1;              // 0..3 : 32-row slab
  const int wn   = wave & 1;               // 0..1 : 32-col slab
  const int r    = lane & 15;
  const int koff = (lane & 16) ? 8 : 0;    // ISA 16-bit A/B fragment layout

  // staging coordinates
  const int sRow = tid >> 2;               // 0..63
  const int sK   = (tid & 3) * 8;          // 0,8,16,24 (8 consecutive k)
  const int sKn  = tid >> 3;               // 0..31  (NN path: k row)
  const int sNq  = (tid & 7) * 8;          // 0..56  (NN path: 8 consecutive n)

  v8f zero = {0.f, 0.f, 0.f, 0.f, 0.f, 0.f, 0.f, 0.f};
  v8f acc[2][2] = {{zero, zero}, {zero, zero}};   // [mi][ni]

  const int nKt = p.K >> 5;
  for (int kt = 0; kt < nKt; ++kt) {
    const int k0 = kt << 5;

    // ---- stage A tile (128x32): 2 rows/thread, b128 load -> b128 store ----
#pragma unroll
    for (int h = 0; h < 2; ++h) {
      const int row = sRow + h * 64;
      const float* ap = A + (long long)(mtile + row) * p.lda + (k0 + sK);
      float4 f0 = *(const float4*)ap;
      float4 f1 = *(const float4*)(ap + 4);
      stage8(f0, f1, &As[row * LDSS + sK]);
      if (kt + 1 < nKt) __builtin_prefetch(ap + 32, 0, 0);
    }
    // ---- stage B tile (64 n x 32 k) transposed to [n][k] ----
    if (p.nT) {   // B stored [N,K]: contiguous in k, same pattern as A
      const float* bp = Bm + (long long)(ntile + sRow) * p.ldb + (k0 + sK);
      float4 f0 = *(const float4*)bp;
      float4 f1 = *(const float4*)(bp + 4);
      stage8(f0, f1, &Bs[sRow * LDSS + sK]);
      if (kt + 1 < nKt) __builtin_prefetch(bp + 32, 0, 0);
    } else {      // B stored [K,N]: coalesced along n, strided b16 scatter
      const float* bp = Bm + (long long)(k0 + sKn) * p.ldb + (ntile + sNq);
      float4 f0 = *(const float4*)bp;
      float4 f1 = *(const float4*)(bp + 4);
      float f[8] = {f0.x, f0.y, f0.z, f0.w, f1.x, f1.y, f1.z, f1.w};
#pragma unroll
      for (int j = 0; j < 8; ++j)
        Bs[(sNq + j) * LDSS + sKn] = (__bf16)f[j];
      if (kt + 1 < nKt) __builtin_prefetch(bp + (long long)32 * p.ldb, 0, 0);
    }
    __syncthreads();

    // ---- fragments (elem e -> K=(e<8?e:e+8)+koff => two b128 LDS reads) ----
    v16bf bfr[2];
#pragma unroll
    for (int ni = 0; ni < 2; ++ni) {
      const __bf16* bp = &Bs[(wn * 32 + ni * 16 + r) * LDSS];
#pragma unroll
      for (int e = 0; e < 16; ++e)
        bfr[ni][e] = bp[(e < 8 ? e : e + 8) + koff];
    }
#pragma unroll
    for (int mi = 0; mi < 2; ++mi) {
      v16bf af;
      const __bf16* ap = &As[(wm * 32 + mi * 16 + r) * LDSS];
#pragma unroll
      for (int e = 0; e < 16; ++e)
        af[e] = ap[(e < 8 ? e : e + 8) + koff];
      acc[mi][0] = __builtin_amdgcn_wmma_f32_16x16x32_bf16(
          false, af, false, bfr[0], (short)0, acc[mi][0], false, false);
      acc[mi][1] = __builtin_amdgcn_wmma_f32_16x16x32_bf16(
          false, af, false, bfr[1], (short)0, acc[mi][1], false, false);
    }
    __syncthreads();
  }

  // ---- epilogue + store (C/D layout: VGPR v -> M, lane -> N) ----
#pragma unroll
  for (int mi = 0; mi < 2; ++mi) {
    const int gm0 = mtile + wm * 32 + mi * 16 + ((lane & 16) ? 8 : 0);
#pragma unroll
    for (int ni = 0; ni < 2; ++ni) {
      const int gn = ntile + wn * 32 + ni * 16 + (lane & 15);
      const float bv = (p.flags & EF_BIAS) ? bias[gn] : 0.f;
#pragma unroll
      for (int v = 0; v < 8; ++v) {
        const int gm = gm0 + v;
        float val = acc[mi][ni][v] * p.alpha + bv;
        if (p.flags & EF_MASK) { if (gn > gm) val = -3.4028234663852886e38f; }
        if (p.flags & EF_GELU) {
          float x = val;
          val = 0.5f * x * (1.f + tanhf(0.7978845608028654f * (x + 0.044715f * x * x * x)));
        }
        if (p.flags & EF_RES) val += res[(long long)gm * p.ldc + gn];
        float* cp = &C[(long long)gm * p.ldc + gn];
        if (p.flags & EF_NT) __builtin_nontemporal_store(val, cp);
        else                 *cp = val;
      }
    }
  }
}

// ---------------------------------------------------------------------------
// Small helper kernels
// ---------------------------------------------------------------------------
__global__ __launch_bounds__(256) void embed_kernel(
    const int* ids, const float* emb, float* x, long long total, int d) {
  long long idx = (long long)blockIdx.x * 256 + threadIdx.x;
  if (idx >= total) return;
  int dd = (int)(idx % d);
  long long tkn = idx / d;
  x[idx] = emb[(long long)ids[tkn] * d + dd];
}

__global__ __launch_bounds__(256) void rmsnorm_kernel(
    const float* x, const float* w, float* y,
    int ncols, int rowsPerW, long long wStride) {
  __shared__ float red[256];
  const long long row = blockIdx.x;
  const float* xr = x + row * ncols;
  float*       yr = y + row * ncols;
  const float* wr = w + (row / rowsPerW) * wStride;
  float s = 0.f;
  for (int c = threadIdx.x; c < ncols; c += 256) { float v = xr[c]; s += v * v; }
  red[threadIdx.x] = s; __syncthreads();
  for (int o = 128; o > 0; o >>= 1) {
    if (threadIdx.x < o) red[threadIdx.x] += red[threadIdx.x + o];
    __syncthreads();
  }
  const float inv = rsqrtf(red[0] / (float)ncols + 1e-6f);
  for (int c = threadIdx.x; c < ncols; c += 256) yr[c] = xr[c] * inv * wr[c];
}

// RoPE in place on q and k; rows enumerate (outer..., t); hd fixed 64
__global__ __launch_bounds__(256) void rope_kernel(
    float* q, float* k, long long nrows, int T, int nh) {
  long long idx = (long long)blockIdx.x * 256 + threadIdx.x;
  long long total = nrows * (long long)nh * 32;
  if (idx >= total) return;
  int i = (int)(idx & 31);
  long long t2 = idx >> 5;
  int h = (int)(t2 % nh);
  long long row = t2 / nh;
  int t = (int)(row % T);
  float inv = __powf(10000.f, -(float)(2 * i) / 64.f);
  float f = (float)t * inv;
  float c = __cosf(f), s = __sinf(f);
  long long base = (row * nh + h) * 64 + i;
  float a1 = q[base], a2 = q[base + 32];
  q[base] = a1 * c - a2 * s; q[base + 32] = a2 * c + a1 * s;
  float b1 = k[base], b2 = k[base + 32];
  k[base] = b1 * c - b2 * s; k[base + 32] = b2 * c + b1 * s;
}

__global__ __launch_bounds__(256) void softmax_kernel(float* sm, int n) {
  __shared__ float red[256];
  float* sr = sm + (long long)blockIdx.x * n;
  float mx = -3.4e38f;
  for (int c = threadIdx.x; c < n; c += 256) mx = fmaxf(mx, sr[c]);
  red[threadIdx.x] = mx; __syncthreads();
  for (int o = 128; o > 0; o >>= 1) {
    if (threadIdx.x < o) red[threadIdx.x] = fmaxf(red[threadIdx.x], red[threadIdx.x + o]);
    __syncthreads();
  }
  mx = red[0]; __syncthreads();
  float sum = 0.f;
  for (int c = threadIdx.x; c < n; c += 256) {
    float e = __expf(sr[c] - mx); sr[c] = e; sum += e;
  }
  red[threadIdx.x] = sum; __syncthreads();
  for (int o = 128; o > 0; o >>= 1) {
    if (threadIdx.x < o) red[threadIdx.x] += red[threadIdx.x + o];
    __syncthreads();
  }
  const float inv = 1.f / red[0];
  for (int c = threadIdx.x; c < n; c += 256) sr[c] *= inv;
}

// cross_merge attention: per (b,t,h,i) -> 4x4 softmax over columns j
__global__ __launch_bounds__(256) void merge_attn_kernel(
    const float* q, const float* k, const float* v, float* o,
    int Bb, int T, int nh, int dcv) {
  long long idx = (long long)blockIdx.x * 256 + threadIdx.x;
  long long total = (long long)Bb * T * nh * 4;
  if (idx >= total) return;
  int i = (int)(idx & 3);
  long long r = idx >> 2;
  int h = (int)(r % nh); r /= nh;
  int t = (int)(r % T);
  int b = (int)(r / T);
  const long long sc = (long long)Bb * T * dcv;            // per-column stride
  const long long base = (((long long)b * T + t) * dcv) + h * 64;
  const float* qi = q + i * sc + base;
  float s[4]; float mx = -3.4e38f;
#pragma unroll
  for (int j = 0; j < 4; ++j) {
    const float* kj = k + j * sc + base;
    float d = 0.f;
    for (int e = 0; e < 64; ++e) d += qi[e] * kj[e];
    d *= 0.125f; s[j] = d; mx = fmaxf(mx, d);
  }
  float sum = 0.f;
#pragma unroll
  for (int j = 0; j < 4; ++j) { s[j] = __expf(s[j] - mx); sum += s[j]; }
  const float inv = 1.f / sum;
  float* oi = o + i * sc + base;
  for (int e = 0; e < 64; ++e) {
    float a = 0.f;
#pragma unroll
    for (int j = 0; j < 4; ++j) a += s[j] * v[j * sc + base + e];
    oi[e] = a * inv;
  }
}

// comb[b,t,c*DC+e] = cs[c,b,t,e]
__global__ __launch_bounds__(256) void permute_kernel(
    const float* cs, float* comb, int Bb, int T, int nc, int dcv) {
  long long idx = (long long)blockIdx.x * 256 + threadIdx.x;
  long long total = (long long)Bb * T * nc * dcv;
  if (idx >= total) return;
  int e = (int)(idx % dcv); long long r = idx / dcv;
  int c = (int)(r % nc); r /= nc;
  int t = (int)(r % T);
  int b = (int)(r / T);
  comb[((((long long)b * T + t) * nc) + c) * dcv + e] =
      cs[((((long long)c * Bb + b) * T) + t) * dcv + e];
}

// ---------------------------------------------------------------------------
// Host side
// ---------------------------------------------------------------------------
static void run_gemm(hipStream_t st,
                     const float* A, const float* Bm, float* C,
                     int M, int N, int K, int lda, int ldb, int ldc,
                     int NB, int NH,
                     long long sAb, long long sAh,
                     long long sBb, long long sBh,
                     long long sCb, long long sCh,
                     int nT, float alpha, int flags,
                     const float* bias, long long biasSb, const float* res) {
  GemmP p;
  p.A = A; p.Bm = Bm; p.C = C; p.bias = bias; p.res = res;
  p.M = M; p.N = N; p.K = K; p.lda = lda; p.ldb = ldb; p.ldc = ldc;
  p.sAb = sAb; p.sAh = sAh; p.sBb = sBb; p.sBh = sBh; p.sCb = sCb; p.sCh = sCh;
  p.biasSb = biasSb; p.NH = NH; p.nT = nT; p.alpha = alpha; p.flags = flags;
  dim3 g((N + TN - 1) / TN, (M + TM - 1) / TM, NB * NH);
  gemm_kernel<<<g, dim3(256), 0, st>>>(p);
}

extern "C" void kernel_launch(void* const* d_in, const int* in_sizes, int n_in,
                              void* d_out, int out_size, void* d_ws, size_t ws_size,
                              hipStream_t stream) {
  (void)in_sizes; (void)n_in; (void)out_size; (void)ws_size;
  const int*   ids       = (const int*)  d_in[0];
  const float* tok_emb   = (const float*)d_in[1];
  const float* trunk_wq  = (const float*)d_in[2];
  const float* trunk_wk  = (const float*)d_in[3];
  const float* trunk_wv  = (const float*)d_in[4];
  const float* trunk_wo  = (const float*)d_in[5];
  const float* trunk_w1  = (const float*)d_in[6];
  const float* trunk_w2  = (const float*)d_in[7];
  const float* trunk_n1  = (const float*)d_in[8];
  const float* trunk_n2  = (const float*)d_in[9];
  const float* col_in_w  = (const float*)d_in[10];
  const float* col_in_b  = (const float*)d_in[11];
  const float* col_wq    = (const float*)d_in[12];
  const float* col_wk    = (const float*)d_in[13];
  const float* col_wv    = (const float*)d_in[14];
  const float* col_wo    = (const float*)d_in[15];
  const float* col_w1    = (const float*)d_in[16];
  const float* col_w2    = (const float*)d_in[17];
  const float* col_n1    = (const float*)d_in[18];
  const float* col_n2    = (const float*)d_in[19];
  const float* mrg_wq    = (const float*)d_in[20];
  const float* mrg_wk    = (const float*)d_in[21];
  const float* mrg_wv    = (const float*)d_in[22];
  const float* mrg_wo    = (const float*)d_in[23];
  const float* mrg_n     = (const float*)d_in[24];
  const float* out_proj  = (const float*)d_in[25];
  const float* final_nrm = (const float*)d_in[26];
  float* out = (float*)d_out;

  const long long NTOK = (long long)BB * TT;  // 2048
  float* ws = (float*)d_ws;
  long long off = 0;
  auto carve = [&](long long n) { float* p = ws + off; off += n; return p; };
  float* xb    = carve(NTOK * DD);                     // trunk activations
  float* hb    = carve(NTOK * DD);                     // norm scratch
  float* qb    = carve(NTOK * DD);
  float* kb    = carve(NTOK * DD);
  float* vb    = carve(NTOK * DD);
  float* ob    = carve(NTOK * DD);
  float* csb   = carve((long long)CCOL * NTOK * DC);
  float* combb = carve(NTOK * DD);
  float* ffb   = carve(NTOK * DFF);                    // == C*B*T*DFFC
  float* sb    = carve((long long)BB * HT * TT * TT);  // == C*B*HC*T*T

  // ---- embedding gather ----
  {
    long long total = NTOK * DD;
    embed_kernel<<<dim3((unsigned)((total + 255) / 256)), dim3(256), 0, stream>>>(
        ids, tok_emb, xb, total, DD);
  }

  // ---- trunk layers ----
  for (int l = 0; l < LT; ++l) {
    const float* wq = trunk_wq + (long long)l * DD * DD;
    const float* wk = trunk_wk + (long long)l * DD * DD;
    const float* wv = trunk_wv + (long long)l * DD * DD;
    const float* wo = trunk_wo + (long long)l * DD * DD;
    const float* w1 = trunk_w1 + (long long)l * DD * DFF;
    const float* w2 = trunk_w2 + (long long)l * DFF * DD;
    const float* n1 = trunk_n1 + (long long)l * DD;
    const float* n2 = trunk_n2 + (long long)l * DD;

    rmsnorm_kernel<<<dim3((unsigned)NTOK), dim3(256), 0, stream>>>(
        xb, n1, hb, DD, (int)NTOK, 0);
    run_gemm(stream, hb, wq, qb, (int)NTOK, DD, DD, DD, DD, DD, 1, 1,
             0,0,0,0,0,0, 0, 1.f, 0, nullptr, 0, nullptr);
    run_gemm(stream, hb, wk, kb, (int)NTOK, DD, DD, DD, DD, DD, 1, 1,
             0,0,0,0,0,0, 0, 1.f, 0, nullptr, 0, nullptr);
    run_gemm(stream, hb, wv, vb, (int)NTOK, DD, DD, DD, DD, DD, 1, 1,
             0,0,0,0,0,0, 0, 1.f, 0, nullptr, 0, nullptr);
    rope_kernel<<<dim3((unsigned)((NTOK * HT * 32 + 255) / 256)), dim3(256), 0, stream>>>(
        qb, kb, NTOK, TT, HT);
    // S = QK^T * hd^-0.5, causal mask; batch (b,h)
    run_gemm(stream, qb, kb, sb, TT, TT, HD, DD, DD, TT, BB, HT,
             (long long)TT * DD, HD, (long long)TT * DD, HD,
             (long long)HT * TT * TT, (long long)TT * TT,
             1, 0.125f, EF_MASK, nullptr, 0, nullptr);
    softmax_kernel<<<dim3((unsigned)(BB * HT * TT)), dim3(256), 0, stream>>>(sb, TT);
    // O = P @ V
    run_gemm(stream, sb, vb, ob, TT, HD, TT, TT, DD, DD, BB, HT,
             (long long)HT * TT * TT, (long long)TT * TT,
             (long long)TT * DD, HD, (long long)TT * DD, HD,
             0, 1.f, 0, nullptr, 0, nullptr);
    // x = x + O @ wo
    run_gemm(stream, ob, wo, xb, (int)NTOK, DD, DD, DD, DD, DD, 1, 1,
             0,0,0,0,0,0, 0, 1.f, EF_RES, nullptr, 0, xb);
    // MLP
    rmsnorm_kernel<<<dim3((unsigned)NTOK), dim3(256), 0, stream>>>(
        xb, n2, hb, DD, (int)NTOK, 0);
    run_gemm(stream, hb, w1, ffb, (int)NTOK, DFF, DD, DD, DFF, DFF, 1, 1,
             0,0,0,0,0,0, 0, 1.f, EF_GELU, nullptr, 0, nullptr);
    run_gemm(stream, ffb, w2, xb, (int)NTOK, DD, DFF, DFF, DD, DD, 1, 1,
             0,0,0,0,0,0, 0, 1.f, EF_RES, nullptr, 0, xb);
  }

  // ---- column projection: cs[c] = x @ col_in_w[c] + col_in_b[c] ----
  run_gemm(stream, xb, col_in_w, csb, (int)NTOK, DC, DD, DD, DC, DC, CCOL, 1,
           0, 0, (long long)DD * DC, 0, NTOK * DC, 0,
           0, 1.f, EF_BIAS, col_in_b, DC, nullptr);

  // ---- column layers ----
  for (int l = 0; l < LC; ++l) {
    rmsnorm_kernel<<<dim3((unsigned)(CCOL * NTOK)), dim3(256), 0, stream>>>(
        csb, col_n1 + (long long)l * DC, hb, DC, (int)NTOK, (long long)LC * DC);
    const long long wSb = (long long)LC * DC * DC;
    run_gemm(stream, hb, col_wq + (long long)l * DC * DC, qb,
             (int)NTOK, DC, DC, DC, DC, DC, CCOL, 1,
             NTOK * DC, 0, wSb, 0, NTOK * DC, 0, 0, 1.f, 0, nullptr, 0, nullptr);
    run_gemm(stream, hb, col_wk + (long long)l * DC * DC, kb,
             (int)NTOK, DC, DC, DC, DC, DC, CCOL, 1,
             NTOK * DC, 0, wSb, 0, NTOK * DC, 0, 0, 1.f, 0, nullptr, 0, nullptr);
    run_gemm(stream, hb, col_wv + (long long)l * DC * DC, vb,
             (int)NTOK, DC, DC, DC, DC, DC, CCOL, 1,
             NTOK * DC, 0, wSb, 0, NTOK * DC, 0, 0, 1.f, 0, nullptr, 0, nullptr);
    rope_kernel<<<dim3((unsigned)((CCOL * NTOK * HC * 32 + 255) / 256)), dim3(256), 0, stream>>>(
        qb, kb, CCOL * NTOK, TT, HC);
    // batch (c*b, h)
    run_gemm(stream, qb, kb, sb, TT, TT, HD, DC, DC, TT, CCOL * BB, HC,
             (long long)TT * DC, HD, (long long)TT * DC, HD,
             (long long)HC * TT * TT, (long long)TT * TT,
             1, 0.125f, EF_MASK, nullptr, 0, nullptr);
    softmax_kernel<<<dim3((unsigned)(CCOL * BB * HC * TT)), dim3(256), 0, stream>>>(sb, TT);
    run_gemm(stream, sb, vb, ob, TT, HD, TT, TT, DC, DC, CCOL * BB, HC,
             (long long)HC * TT * TT, (long long)TT * TT,
             (long long)TT * DC, HD, (long long)TT * DC, HD,
             0, 1.f, 0, nullptr, 0, nullptr);
    run_gemm(stream, ob, col_wo + (long long)l * DC * DC, csb,
             (int)NTOK, DC, DC, DC, DC, DC, CCOL, 1,
             NTOK * DC, 0, wSb, 0, NTOK * DC, 0, 0, 1.f, EF_RES, nullptr, 0, csb);
    // column MLP
    rmsnorm_kernel<<<dim3((unsigned)(CCOL * NTOK)), dim3(256), 0, stream>>>(
        csb, col_n2 + (long long)l * DC, hb, DC, (int)NTOK, (long long)LC * DC);
    run_gemm(stream, hb, col_w1 + (long long)l * DC * DFFC, ffb,
             (int)NTOK, DFFC, DC, DC, DFFC, DFFC, CCOL, 1,
             NTOK * DC, 0, (long long)LC * DC * DFFC, 0, NTOK * DFFC, 0,
             0, 1.f, EF_GELU, nullptr, 0, nullptr);
    run_gemm(stream, ffb, col_w2 + (long long)l * DFFC * DC, csb,
             (int)NTOK, DC, DFFC, DFFC, DC, DC, CCOL, 1,
             NTOK * DFFC, 0, (long long)LC * DFFC * DC, 0, NTOK * DC, 0,
             0, 1.f, EF_RES, nullptr, 0, csb);

    // ---- cross merge every MERGE_EVERY layers ----
    if ((l + 1) % 2 == 0) {
      const int m = (l + 1) / 2 - 1;
      rmsnorm_kernel<<<dim3((unsigned)(CCOL * NTOK)), dim3(256), 0, stream>>>(
          csb, mrg_n + (long long)m * DC, hb, DC, (int)(CCOL * NTOK), 0);
      run_gemm(stream, hb, mrg_wq + (long long)m * DC * DC, qb,
               (int)(CCOL * NTOK), DC, DC, DC, DC, DC, 1, 1,
               0,0,0,0,0,0, 0, 1.f, 0, nullptr, 0, nullptr);
      run_gemm(stream, hb, mrg_wk + (long long)m * DC * DC, kb,
               (int)(CCOL * NTOK), DC, DC, DC, DC, DC, 1, 1,
               0,0,0,0,0,0, 0, 1.f, 0, nullptr, 0, nullptr);
      run_gemm(stream, hb, mrg_wv + (long long)m * DC * DC, vb,
               (int)(CCOL * NTOK), DC, DC, DC, DC, DC, 1, 1,
               0,0,0,0,0,0, 0, 1.f, 0, nullptr, 0, nullptr);
      merge_attn_kernel<<<dim3((unsigned)((BB * TT * HC * 4 + 255) / 256)), dim3(256), 0, stream>>>(
          qb, kb, vb, ob, BB, TT, HC, DC);
      run_gemm(stream, ob, mrg_wo + (long long)m * DC * DC, csb,
               (int)(CCOL * NTOK), DC, DC, DC, DC, DC, 1, 1,
               0,0,0,0,0,0, 0, 1.f, EF_RES, nullptr, 0, csb);
    }
  }

  // ---- output head ----
  permute_kernel<<<dim3((unsigned)((NTOK * DD + 255) / 256)), dim3(256), 0, stream>>>(
      csb, combb, BB, TT, CCOL, DC);
  run_gemm(stream, combb, out_proj, hb, (int)NTOK, DD, DD, DD, DD, DD, 1, 1,
           0,0,0,0,0,0, 0, 1.f, 0, nullptr, 0, nullptr);
  rmsnorm_kernel<<<dim3((unsigned)NTOK), dim3(256), 0, stream>>>(
      hb, final_nrm, combb, DD, (int)NTOK, 0);
  // logits = comb @ tok_emb^T  (NT gemm, N = vocab, nontemporal store)
  run_gemm(stream, combb, tok_emb, out, (int)NTOK, VV, DD, DD, DD, VV, 1, 1,
           0,0,0,0,0,0, 1, 1.f, EF_NT, nullptr, 0, nullptr);
}
